// AttentionEncoder_1700807049676
// MI455X (gfx1250) — matrix operand order repaired and checked by reading, not compiled, not run on hardware
//
#include <hip/hip_runtime.h>

// ---------------------------------------------------------------------------
// Types for CDNA5 WMMA (wave32)
// ---------------------------------------------------------------------------
typedef __attribute__((ext_vector_type(16))) __bf16 v16bf;
typedef __attribute__((ext_vector_type(8)))  __bf16 v8bf;
typedef __attribute__((ext_vector_type(8)))  float  v8f;

#define N_NODES   50000
#define N_HEDGES  50000
#define NNZ_E     1600000
#define IN_CH     256
#define OUT_CH    256
#define NODE_CH   128
#define Q_CH      64
#define NEG_SLOPE 0.2f
#define SCALE_Q   0.125f   // 1/sqrt(64)

static inline int ceildiv(long long a, long long b) { return (int)((a + b - 1) / b); }

// ---------------------------------------------------------------------------
// Helpers: order-preserving float <-> uint for atomicMax-based segment max
// ---------------------------------------------------------------------------
__device__ __forceinline__ unsigned f2ord(float f) {
  unsigned u = __float_as_uint(f);
  return (u & 0x80000000u) ? ~u : (u | 0x80000000u);
}
__device__ __forceinline__ float ord2f(unsigned o) {
  return (o & 0x80000000u) ? __uint_as_float(o & 0x7FFFFFFFu)
                           : __uint_as_float(~o);
}

// ---------------------------------------------------------------------------
// Elementwise utility kernels
// ---------------------------------------------------------------------------
__global__ void fill_u32_kernel(unsigned* __restrict__ p, unsigned v, int n) {
  int i = blockIdx.x * blockDim.x + threadIdx.x;
  if (i < n) p[i] = v;
}

__global__ void f32_to_bf16_kernel(const float* __restrict__ x, __bf16* __restrict__ y, int n) {
  int i = blockIdx.x * blockDim.x + threadIdx.x;
  if (i < n) y[i] = (__bf16)x[i];
}

// w: [K,N] f32 row-major  ->  wt: [N,K] bf16 row-major (transposed)
__global__ void transpose_bf16_kernel(const float* __restrict__ w, __bf16* __restrict__ wt,
                                      int K, int N) {
  int i = blockIdx.x * blockDim.x + threadIdx.x;
  if (i < K * N) {
    int k = i / N, n = i % N;
    wt[(size_t)n * K + k] = (__bf16)w[i];
  }
}

// ---------------------------------------------------------------------------
// bf16 WMMA GEMM:  C[M,N] (f32) = A[M,K] (bf16, row-major) * BT[N,K]^T (bf16)
// One wave (32 threads) computes a 16x64 output tile (4 WMMA accumulators).
// Requires: M % 16 == 0, K % 32 == 0, N % 64 == 0.   (All true here.)
// A fragment (contiguous k, same as B): lanes 0-15 rows M, K chunks
// {0..7,8..15}; lanes 16-31 same rows, K chunks {16..23,24..31}.
// B fragment: lane holds column n = lane&15, K = 16*(lane>=16) + 0..15,
// contiguous in BT row -> single 32-byte load.
// ---------------------------------------------------------------------------
__global__ void gemm_bf16_wmma_kernel(const __bf16* __restrict__ A,
                                      const __bf16* __restrict__ BT,
                                      float* __restrict__ C, int K, int N) {
  const int lane = threadIdx.x & 31;
  const int lcol = lane & 15;
  const int lhi  = lane >> 4;                 // 0 or 1
  const size_t m0 = (size_t)blockIdx.y * 16;
  const int n0 = blockIdx.x * 64;

  const __bf16* arow = A + (m0 + (size_t)lcol) * K;
  const int aoff = lhi * 16;                  // K offset of first 8-elem chunk
  const __bf16* brow0 = BT + ((size_t)(n0 +  0 + lcol)) * K + lhi * 16;
  const __bf16* brow1 = BT + ((size_t)(n0 + 16 + lcol)) * K + lhi * 16;
  const __bf16* brow2 = BT + ((size_t)(n0 + 32 + lcol)) * K + lhi * 16;
  const __bf16* brow3 = BT + ((size_t)(n0 + 48 + lcol)) * K + lhi * 16;

  v8f acc0 = {}, acc1 = {}, acc2 = {}, acc3 = {};

  for (int k0 = 0; k0 < K; k0 += 32) {
    v8bf alo = *(const v8bf*)(arow + k0 + aoff);        // K = k0 + aoff + 0..7
    v8bf ahi = *(const v8bf*)(arow + k0 + 8 + aoff);    // K = k0 + aoff + 8..15
    v16bf a;
#pragma unroll
    for (int i = 0; i < 8; ++i) { a[i] = alo[i]; a[i + 8] = ahi[i]; }

    v16bf b0 = *(const v16bf*)(brow0 + k0);
    v16bf b1 = *(const v16bf*)(brow1 + k0);
    v16bf b2 = *(const v16bf*)(brow2 + k0);
    v16bf b3 = *(const v16bf*)(brow3 + k0);

    acc0 = __builtin_amdgcn_wmma_f32_16x16x32_bf16(false, a, false, b0, (short)0, acc0, false, false);
    acc1 = __builtin_amdgcn_wmma_f32_16x16x32_bf16(false, a, false, b1, (short)0, acc1, false, false);
    acc2 = __builtin_amdgcn_wmma_f32_16x16x32_bf16(false, a, false, b2, (short)0, acc2, false, false);
    acc3 = __builtin_amdgcn_wmma_f32_16x16x32_bf16(false, a, false, b3, (short)0, acc3, false, false);
  }

  // D layout: VGPR r, lane -> C[m0 + r + 8*lhi][n0 + (lane&15) + 16*t]
  float* crow = C + (m0 + (size_t)lhi * 8) * N + n0 + lcol;
#pragma unroll
  for (int r = 0; r < 8; ++r) {
    crow[(size_t)r * N +  0] = acc0[r];
    crow[(size_t)r * N + 16] = acc1[r];
    crow[(size_t)r * N + 32] = acc2[r];
    crow[(size_t)r * N + 48] = acc3[r];
  }
}

// ---------------------------------------------------------------------------
// Edge pass 1: score[e] = leaky_relu( dot_64(fa[ia[e]], fb[ib[e]]) )
//              segmax[iseg[e]] = max(segmax, ord(score))   (atomic)
// ---------------------------------------------------------------------------
__global__ void edge_score_kernel(const float* __restrict__ fa, const long long* __restrict__ ia,
                                  const float* __restrict__ fb, const long long* __restrict__ ib,
                                  const long long* __restrict__ iseg,
                                  float* __restrict__ score, unsigned* __restrict__ segmax,
                                  int nE) {
  int e = blockIdx.x * blockDim.x + threadIdx.x;
  if (e >= nE) return;
  const float4* ra = (const float4*)(fa + (size_t)ia[e] * Q_CH);
  const float4* rb = (const float4*)(fb + (size_t)ib[e] * Q_CH);
  float d = 0.f;
#pragma unroll
  for (int i = 0; i < Q_CH / 4; ++i) {
    float4 a = ra[i], b = rb[i];
    d += a.x * b.x + a.y * b.y + a.z * b.z + a.w * b.w;
  }
  float s = d > 0.f ? d : NEG_SLOPE * d;
  score[e] = s;
  atomicMax(segmax + iseg[e], f2ord(s));
}

// ---------------------------------------------------------------------------
// Edge pass 2: ex[e] = exp(score[e] - max[seg]); segsum[seg] += ex (atomic)
// ---------------------------------------------------------------------------
__global__ void edge_expsum_kernel(const float* __restrict__ score,
                                   const unsigned* __restrict__ segmax,
                                   const long long* __restrict__ iseg,
                                   float* __restrict__ exv, float* __restrict__ segsum,
                                   int nE) {
  int e = blockIdx.x * blockDim.x + threadIdx.x;
  if (e >= nE) return;
  long long sg = iseg[e];
  float m = ord2f(segmax[sg]);
  float ex = __expf(score[e] - m);
  exv[e] = ex;
  atomicAdd(segsum + sg, ex);
}

// ---------------------------------------------------------------------------
// Edge pass 3: out[iseg[e], :] += scale * ex[e]/(sum+1e-16) * feat[isrc[e], :]
// One wave per edge; per-channel f32 atomics (L2-resident target rows).
// ---------------------------------------------------------------------------
__global__ void edge_scatter_kernel(const float* __restrict__ exv,
                                    const float* __restrict__ segsum,
                                    const long long* __restrict__ iseg,
                                    const long long* __restrict__ isrc,
                                    const float* __restrict__ feat,
                                    float* __restrict__ out,
                                    int C, int nE) {
  int e = (blockIdx.x * blockDim.x + threadIdx.x) >> 5;
  int lane = threadIdx.x & 31;
  if (e >= nE) return;
  long long sg = iseg[e], sr = isrc[e];
  float coeff = SCALE_Q * exv[e] / (segsum[sg] + 1e-16f);
  const float* f = feat + (size_t)sr * C;
  float*       o = out  + (size_t)sg * C;
  for (int c = lane; c < C; c += 32) atomicAdd(o + c, coeff * f[c]);
}

// ---------------------------------------------------------------------------
// Host launcher
// ---------------------------------------------------------------------------
extern "C" void kernel_launch(void* const* d_in, const int* in_sizes, int n_in,
                              void* d_out, int out_size, void* d_ws, size_t ws_size,
                              hipStream_t stream) {
  (void)in_sizes; (void)n_in; (void)out_size;

  const float*     p    = (const float*)d_in[0];     // [N, 128]
  const float*     q    = (const float*)d_in[1];     // [H, 256]
  const long long* hidx = (const long long*)d_in[2]; // [2, E] int64
  const float*     w1   = (const float*)d_in[3];     // [256,128]
  const float*     w2   = (const float*)d_in[4];     // [256, 64]
  const float*     w3   = (const float*)d_in[5];     // [128, 64]
  const float*     w4   = (const float*)d_in[6];     // [128,256]
  const float*     w5   = (const float*)d_in[7];     // [128, 64]
  const float*     w6   = (const float*)d_in[8];     // [256, 64]
  float* q_out = (float*)d_out;                      // [H, 256]

  const long long* node_i  = hidx;          // [E]
  const long long* hedge_j = hidx + NNZ_E;  // [E]

  // ---- bump allocator over workspace ----
  char* wsp = (char*)d_ws;
  size_t off = 0;
  auto alloc = [&](size_t bytes) -> void* {
    void* r = wsp + off;
    off = (off + bytes + 255) & ~(size_t)255;
    return r;
  };

  __bf16* q_bf    = (__bf16*)alloc((size_t)N_HEDGES * IN_CH   * 2);
  __bf16* p_bf    = (__bf16*)alloc((size_t)N_NODES  * NODE_CH * 2);
  __bf16* pout_bf = (__bf16*)alloc((size_t)N_NODES  * NODE_CH * 2);
  __bf16* w1T = (__bf16*)alloc((size_t)NODE_CH * IN_CH   * 2);
  __bf16* w2T = (__bf16*)alloc((size_t)Q_CH    * IN_CH   * 2);
  __bf16* w3T = (__bf16*)alloc((size_t)Q_CH    * NODE_CH * 2);
  __bf16* w4T = (__bf16*)alloc((size_t)OUT_CH  * NODE_CH * 2);
  __bf16* w5T = (__bf16*)alloc((size_t)Q_CH    * NODE_CH * 2);
  __bf16* w6T = (__bf16*)alloc((size_t)Q_CH    * IN_CH   * 2);
  float* q_p   = (float*)alloc((size_t)N_HEDGES * NODE_CH * 4);
  float* q_pp  = (float*)alloc((size_t)N_HEDGES * Q_CH    * 4);
  float* q_ppp = (float*)alloc((size_t)N_NODES  * Q_CH    * 4);
  float* p_out = (float*)alloc((size_t)N_NODES  * NODE_CH * 4);
  float* p_p   = (float*)alloc((size_t)N_NODES  * OUT_CH  * 4);
  float* p_pp  = (float*)alloc((size_t)N_NODES  * Q_CH    * 4);
  float* p_ppp = (float*)alloc((size_t)N_HEDGES * Q_CH    * 4);
  float*    score  = (float*)alloc((size_t)NNZ_E * 4);
  float*    exv    = (float*)alloc((size_t)NNZ_E * 4);
  unsigned* segmax = (unsigned*)alloc((size_t)N_NODES * 4);
  float*    segsum = (float*)alloc((size_t)N_NODES * 4);

  if (off > ws_size) return;  // workspace too small: do nothing (no OOB writes)

  const int TB = 256;
  auto g1 = [&](long long n) { return dim3((unsigned)ceildiv(n, TB)); };

  // ---- stage 0: precision conversion + weight transposition ----
  hipLaunchKernelGGL(f32_to_bf16_kernel, g1((long long)N_HEDGES * IN_CH), dim3(TB), 0, stream,
                     q, q_bf, N_HEDGES * IN_CH);
  hipLaunchKernelGGL(f32_to_bf16_kernel, g1((long long)N_NODES * NODE_CH), dim3(TB), 0, stream,
                     p, p_bf, N_NODES * NODE_CH);
  hipLaunchKernelGGL(transpose_bf16_kernel, g1(IN_CH * NODE_CH), dim3(TB), 0, stream, w1, w1T, IN_CH, NODE_CH);
  hipLaunchKernelGGL(transpose_bf16_kernel, g1(IN_CH * Q_CH),    dim3(TB), 0, stream, w2, w2T, IN_CH, Q_CH);
  hipLaunchKernelGGL(transpose_bf16_kernel, g1(NODE_CH * Q_CH),  dim3(TB), 0, stream, w3, w3T, NODE_CH, Q_CH);
  hipLaunchKernelGGL(transpose_bf16_kernel, g1(NODE_CH * OUT_CH),dim3(TB), 0, stream, w4, w4T, NODE_CH, OUT_CH);
  hipLaunchKernelGGL(transpose_bf16_kernel, g1(NODE_CH * Q_CH),  dim3(TB), 0, stream, w5, w5T, NODE_CH, Q_CH);
  hipLaunchKernelGGL(transpose_bf16_kernel, g1(IN_CH * Q_CH),    dim3(TB), 0, stream, w6, w6T, IN_CH, Q_CH);

  // ---- stage 1 GEMMs (WMMA): q_p = q@w1, q_pp = q@w2, q_ppp = p@w3 ----
  hipLaunchKernelGGL(gemm_bf16_wmma_kernel, dim3(NODE_CH / 64, N_HEDGES / 16), dim3(32), 0, stream,
                     q_bf, w1T, q_p, IN_CH, NODE_CH);
  hipLaunchKernelGGL(gemm_bf16_wmma_kernel, dim3(Q_CH / 64, N_HEDGES / 16), dim3(32), 0, stream,
                     q_bf, w2T, q_pp, IN_CH, Q_CH);
  hipLaunchKernelGGL(gemm_bf16_wmma_kernel, dim3(Q_CH / 64, N_NODES / 16), dim3(32), 0, stream,
                     p_bf, w3T, q_ppp, NODE_CH, Q_CH);

  // ---- stage 1 attention: segments = node_i ----
  hipLaunchKernelGGL(fill_u32_kernel, g1(N_NODES), dim3(TB), 0, stream, segmax, 0u, N_NODES);
  hipLaunchKernelGGL(fill_u32_kernel, g1(N_NODES), dim3(TB), 0, stream, (unsigned*)segsum, 0u, N_NODES);
  hipLaunchKernelGGL(fill_u32_kernel, g1((long long)N_NODES * NODE_CH), dim3(TB), 0, stream,
                     (unsigned*)p_out, 0u, N_NODES * NODE_CH);
  hipLaunchKernelGGL(edge_score_kernel, g1(NNZ_E), dim3(TB), 0, stream,
                     q_ppp, node_i, q_pp, hedge_j, node_i, score, segmax, NNZ_E);
  hipLaunchKernelGGL(edge_expsum_kernel, g1(NNZ_E), dim3(TB), 0, stream,
                     score, segmax, node_i, exv, segsum, NNZ_E);
  hipLaunchKernelGGL(edge_scatter_kernel, g1((long long)NNZ_E * 32), dim3(TB), 0, stream,
                     exv, segsum, node_i, hedge_j, q_p, p_out, NODE_CH, NNZ_E);

  // ---- stage 2 GEMMs: p_p = p_out@w4, p_pp = p_out@w5, p_ppp = q@w6 ----
  hipLaunchKernelGGL(f32_to_bf16_kernel, g1((long long)N_NODES * NODE_CH), dim3(TB), 0, stream,
                     p_out, pout_bf, N_NODES * NODE_CH);
  hipLaunchKernelGGL(gemm_bf16_wmma_kernel, dim3(OUT_CH / 64, N_NODES / 16), dim3(32), 0, stream,
                     pout_bf, w4T, p_p, NODE_CH, OUT_CH);
  hipLaunchKernelGGL(gemm_bf16_wmma_kernel, dim3(Q_CH / 64, N_NODES / 16), dim3(32), 0, stream,
                     pout_bf, w5T, p_pp, NODE_CH, Q_CH);
  hipLaunchKernelGGL(gemm_bf16_wmma_kernel, dim3(Q_CH / 64, N_HEDGES / 16), dim3(32), 0, stream,
                     q_bf, w6T, p_ppp, IN_CH, Q_CH);

  // ---- stage 2 attention: segments = hedge_j; output = q_out ----
  hipLaunchKernelGGL(fill_u32_kernel, g1(N_HEDGES), dim3(TB), 0, stream, segmax, 0u, N_HEDGES);
  hipLaunchKernelGGL(fill_u32_kernel, g1(N_HEDGES), dim3(TB), 0, stream, (unsigned*)segsum, 0u, N_HEDGES);
  hipLaunchKernelGGL(fill_u32_kernel, g1((long long)N_HEDGES * OUT_CH), dim3(TB), 0, stream,
                     (unsigned*)q_out, 0u, N_HEDGES * OUT_CH);
  hipLaunchKernelGGL(edge_score_kernel, g1(NNZ_E), dim3(TB), 0, stream,
                     p_pp, node_i, p_ppp, hedge_j, hedge_j, score, segmax, NNZ_E);
  hipLaunchKernelGGL(edge_expsum_kernel, g1(NNZ_E), dim3(TB), 0, stream,
                     score, segmax, hedge_j, exv, segsum, NNZ_E);
  hipLaunchKernelGGL(edge_scatter_kernel, g1((long long)NNZ_E * 32), dim3(TB), 0, stream,
                     exv, segsum, hedge_j, node_i, p_p, q_out, OUT_CH, NNZ_E);
}
